// Encoder_19387482374826
// MI455X (gfx1250) — matrix operand order repaired and checked by reading, not compiled, not run on hardware
//
#include <hip/hip_runtime.h>
#include <hip/hip_bf16.h>

// ---------------------------------------------------------------------------
// DA-RNN input-attention encoder for MI455X (gfx1250, wave32, WMMA).
// B=512, T-1=63, N=512, H=256.
//
// Per-step gates GEMM  C[512,1024] = [wx|h](512x768, bf16) @ Wcat^T(768x1024, bf16)
// runs on v_wmma_f32_16x16x32_bf16 (f32 accumulate), staged into LDS via
// GLOBAL_LOAD_ASYNC_TO_LDS_B128 (ASYNCcnt) when the builtin is available.
// The tanh attention loop (~1e9 tanh) uses v_tanh_f32 when available.
// ---------------------------------------------------------------------------

#define BB   512
#define TM1  63
#define NN   512
#define HH   256
#define H4   1024
#define KC   768            // N + H  (concat GEMM K)
#define LOG2E 1.4426950408889634f

typedef __bf16 v16bf __attribute__((ext_vector_type(16)));
typedef float  v8f   __attribute__((ext_vector_type(8)));
typedef int    v4i   __attribute__((ext_vector_type(4)));

union Frag16 { v16bf v; uint4 u[2]; };

#define AS1 __attribute__((address_space(1)))
#define AS3 __attribute__((address_space(3)))

// ---- CDNA5 async global->LDS copy (ASYNCcnt path), guarded ----------------
#if defined(__HIP_DEVICE_COMPILE__) && \
    __has_builtin(__builtin_amdgcn_global_load_async_to_lds_b128)
#define HAVE_ASYNC_LDS 1
__device__ __forceinline__ void async_copy16(__bf16* lds_dst, const __bf16* gsrc) {
  // Builtin signature wants v4i pointers: AS1 (global) src, AS3 (LDS) dst.
  // Generic LDS pointer: LDS offset lives in the low 32 bits (ISA aperture rule).
  __builtin_amdgcn_global_load_async_to_lds_b128(
      (AS1 v4i*)(unsigned long long)gsrc,
      (AS3 v4i*)(unsigned)(unsigned long long)lds_dst,
      0, 0);
}
__device__ __forceinline__ void wait_async0() {
#if __has_builtin(__builtin_amdgcn_s_wait_asynccnt)
  __builtin_amdgcn_s_wait_asynccnt(0);
#else
  asm volatile("s_wait_asynccnt 0" ::: "memory");
#endif
}
#else
#define HAVE_ASYNC_LDS 0
#endif

__device__ __forceinline__ float fexp2(float x) { return __builtin_amdgcn_exp2f(x); }
__device__ __forceinline__ float frcp (float x) { return __builtin_amdgcn_rcpf(x); }
__device__ __forceinline__ float fast_tanh(float x) {
#if defined(__HIP_DEVICE_COMPILE__) && __has_builtin(__builtin_amdgcn_tanhf)
  return __builtin_amdgcn_tanhf(x);            // v_tanh_f32 (single TRANS op)
#else
  float e = fexp2(x * (2.0f * LOG2E));         // tanh = 1 - 2/(exp2(2x*log2e)+1)
  return 1.0f - 2.0f * frcp(e + 1.0f);
#endif
}
__device__ __forceinline__ float fast_sigmoid(float x) {
  return frcp(1.0f + fexp2(-x * LOG2E));
}

// ---------------------------------------------------------------------------
// Setup kernels
// ---------------------------------------------------------------------------

// Wcat^T packed bf16: Wp[col][k], col in [0,1024), k<512 -> W_ih, else W_hh.
__global__ __launch_bounds__(256) void pack_weights(const float* __restrict__ W_ih,
                                                    const float* __restrict__ W_hh,
                                                    __bf16* __restrict__ Wp) {
  int gid = blockIdx.x * 256 + threadIdx.x;          // 1024*768 total
  int col = gid / KC, k = gid - col * KC;
  float v = (k < NN) ? W_ih[col * NN + k] : W_hh[col * HH + (k - NN)];
  Wp[gid] = (__bf16)v;
}

__global__ __launch_bounds__(256) void init_state(float* __restrict__ h,
                                                  float* __restrict__ c,
                                                  __bf16* __restrict__ a_bf) {
  int gid = blockIdx.x * 256 + threadIdx.x;          // B*H total
  h[gid] = 0.0f; c[gid] = 0.0f;
  int b = gid >> 8, j = gid & 255;
  a_bf[b * KC + NN + j] = (__bf16)0.0f;              // h half of concat A
}

// t2[b,n,s] = bd[s] + sum_t x[b,t,n]*Wd[s,t], stored bf16 with stride 64.
__global__ __launch_bounds__(64) void compute_t2(const float* __restrict__ x,
                                                 const float* __restrict__ Wd,
                                                 const float* __restrict__ bd,
                                                 __bf16* __restrict__ t2) {
  int bn = blockIdx.x;                               // b*512 + n
  int b = bn >> 9, n = bn & 511;
  int s = threadIdx.x;
  float acc = 0.0f;
  if (s < TM1) {
    acc = bd[s];
    const float* wr = Wd + s * TM1;                  // 15.9 KB matrix: L2 resident
    const float* xp = x + (size_t)b * TM1 * NN + n;
    #pragma unroll 7
    for (int t = 0; t < TM1; ++t) acc = fmaf(xp[(size_t)t * NN], wr[t], acc);
  }
  t2[(size_t)bn * 64 + s] = (__bf16)acc;             // pad slot 63 with 0
}

// ---------------------------------------------------------------------------
// Per-step kernels
// ---------------------------------------------------------------------------

// t1[b,s] = bc[s] + h[b,:]@Wc[s,0:256] + c[b,:]@Wc[s,256:512]
__global__ __launch_bounds__(256) void step_t1(const float* __restrict__ h,
                                               const float* __restrict__ c,
                                               const float* __restrict__ Wc,
                                               const float* __restrict__ bc,
                                               float* __restrict__ t1) {
  int gid = blockIdx.x * 256 + threadIdx.x;          // B*64
  int b = gid >> 6, s = gid & 63;
  if (s >= TM1) return;
  const float* wr = Wc + (size_t)s * (2 * HH);       // 129 KB: L2 resident, reused 63x
  const float* hb = h + b * HH;
  const float* cb = c + b * HH;
  float acc = bc[s];
  #pragma unroll 8
  for (int j = 0; j < HH; ++j) acc = fmaf(hb[j], wr[j], acc);
  #pragma unroll 8
  for (int j = 0; j < HH; ++j) acc = fmaf(cb[j], wr[HH + j], acc);
  t1[gid] = acc;
}

// Attention + softmax + wx.  One block per batch row; each thread owns 2 n's.
// Hot loop: 2*63 tanh per thread, t2 read as 128-bit bf16 chunks.
__global__ __launch_bounds__(256) void step_attn(const float* __restrict__ x,
                                                 const __bf16* __restrict__ t2,
                                                 const float* __restrict__ t1,
                                                 const float* __restrict__ Wa,
                                                 const float* __restrict__ ba,
                                                 float* __restrict__ out_w,
                                                 float* __restrict__ out_attn,
                                                 __bf16* __restrict__ a_bf,
                                                 int t) {
  __shared__ float ts[TM1];
  __shared__ float wa[TM1];
  __shared__ float red[256];
  const int b = blockIdx.x;
  const int tid = threadIdx.x;
  if (tid < TM1) { ts[tid] = t1[b * 64 + tid]; wa[tid] = Wa[tid]; }
  __syncthreads();

  float sc[2];
  #pragma unroll
  for (int p = 0; p < 2; ++p) {
    const int n = tid + p * 256;
    const __bf16* t2p = t2 + (size_t)(b * NN + n) * 64;   // 128B-aligned row
    float s = ba[0];
    #pragma unroll
    for (int ch = 0; ch < 8; ++ch) {                       // 8 x 16B vector loads
      uint4 d = ((const uint4*)t2p)[ch];
      const __bf16* e = (const __bf16*)&d;
      #pragma unroll
      for (int q = 0; q < 8; ++q) {
        int k = ch * 8 + q;
        if (k < TM1) {
          float z = fast_tanh(ts[k] + (float)e[q]);
          s = fmaf(wa[k], z, s);
        }
      }
    }
    sc[p] = s;
  }

  // block softmax over n = 512
  float m = fmaxf(sc[0], sc[1]);
  red[tid] = m; __syncthreads();
  for (int off = 128; off > 0; off >>= 1) {
    if (tid < off) red[tid] = fmaxf(red[tid], red[tid + off]);
    __syncthreads();
  }
  m = red[0]; __syncthreads();
  float e0 = fexp2((sc[0] - m) * LOG2E);
  float e1 = fexp2((sc[1] - m) * LOG2E);
  red[tid] = e0 + e1; __syncthreads();
  for (int off = 128; off > 0; off >>= 1) {
    if (tid < off) red[tid] += red[tid + off];
    __syncthreads();
  }
  const float rs = frcp(red[0]);

  const size_t xoff = (size_t)b * TM1 * NN + (size_t)t * NN;
  #pragma unroll
  for (int p = 0; p < 2; ++p) {
    const int n = tid + p * 256;
    float attn = (p ? e1 : e0) * rs;
    float wx = attn * x[xoff + n];
    out_attn[xoff + n] = attn;
    out_w[xoff + n] = wx;
    a_bf[b * KC + n] = (__bf16)wx;                   // wx half of concat A
  }
}

// ---------------------------------------------------------------------------
// WMMA gates GEMM: gates[512,1024] = A(512x768 bf16) @ Wp^T (Wp is [1024][768])
// Block tile 128x128, 8 wave32s, each wave 64x32 = 4x2 wmma tiles, K-steps of 32.
// LDS rows padded to 40 bf16 (20 dwords) so 16-lane fragment reads are
// conflict-free across the 64 banks.
// ---------------------------------------------------------------------------
#define GBM 128
#define GBN 128
#define GBK 32
#define LDP 40

__global__ __launch_bounds__(256) void step_gates_wmma(const __bf16* __restrict__ A,
                                                       const __bf16* __restrict__ Wp,
                                                       float* __restrict__ gates) {
  __shared__ __attribute__((aligned(16))) __bf16 lA[GBM * LDP];
  __shared__ __attribute__((aligned(16))) __bf16 lB[GBN * LDP];

  const int m0 = blockIdx.y * GBM;
  const int n0 = blockIdx.x * GBN;
  const int tid = threadIdx.x;
  const int wave = tid >> 5;
  const int lane = tid & 31;
  const int hlf  = lane >> 4;       // which 16-lane half
  const int lrow = lane & 15;
  const int wm = (wave >> 2) * 64;  // 2 waves along M
  const int wn = (wave & 3) * 32;   // 4 waves along N

  const v8f vzero = {0.f, 0.f, 0.f, 0.f, 0.f, 0.f, 0.f, 0.f};
  v8f acc[4][2];
  #pragma unroll
  for (int i = 0; i < 4; ++i)
    #pragma unroll
    for (int j = 0; j < 2; ++j) acc[i][j] = vzero;

  // Cooperative tile staging: a 32-bf16 row is 64B = 4x16B chunks.
  // Thread covers chunk (tid&3) of rows (tid>>2) and (tid>>2)+64.
  const int r0 = tid >> 2;          // 0..63
  const int q0 = tid & 3;           // 0..3  (16B chunk within row)

  for (int k0 = 0; k0 < KC; k0 += GBK) {
#if HAVE_ASYNC_LDS
    async_copy16(lA + (size_t)r0 * LDP + q0 * 8,
                 A  + (size_t)(m0 + r0) * KC + k0 + q0 * 8);
    async_copy16(lA + (size_t)(r0 + 64) * LDP + q0 * 8,
                 A  + (size_t)(m0 + r0 + 64) * KC + k0 + q0 * 8);
    async_copy16(lB + (size_t)r0 * LDP + q0 * 8,
                 Wp + (size_t)(n0 + r0) * KC + k0 + q0 * 8);
    async_copy16(lB + (size_t)(r0 + 64) * LDP + q0 * 8,
                 Wp + (size_t)(n0 + r0 + 64) * KC + k0 + q0 * 8);
    wait_async0();
#else
    *(uint4*)(lA + (size_t)r0 * LDP + q0 * 8) =
        *(const uint4*)(A + (size_t)(m0 + r0) * KC + k0 + q0 * 8);
    *(uint4*)(lA + (size_t)(r0 + 64) * LDP + q0 * 8) =
        *(const uint4*)(A + (size_t)(m0 + r0 + 64) * KC + k0 + q0 * 8);
    *(uint4*)(lB + (size_t)r0 * LDP + q0 * 8) =
        *(const uint4*)(Wp + (size_t)(n0 + r0) * KC + k0 + q0 * 8);
    *(uint4*)(lB + (size_t)(r0 + 64) * LDP + q0 * 8) =
        *(const uint4*)(Wp + (size_t)(n0 + r0 + 64) * KC + k0 + q0 * 8);
#endif
    __syncthreads();

    if (k0 + GBK < KC) {            // pull next K slice toward the WGP caches
      __builtin_prefetch(A  + (size_t)(m0 + r0) * KC + k0 + GBK, 0, 3);
      __builtin_prefetch(Wp + (size_t)(n0 + r0) * KC + k0 + GBK, 0, 3);
    }

    // B fragments: half 0 -> K {0..7,16..23}, half 1 -> K {8..15,24..31} of its column
    Frag16 bf[2];
    #pragma unroll
    for (int j = 0; j < 2; ++j) {
      const __bf16* bp = lB + (size_t)(wn + j * 16 + lrow) * LDP + hlf * 8;
      bf[j].u[0] = ((const uint4*)bp)[0];
      bf[j].u[1] = ((const uint4*)(bp + 16))[0];
    }

    #pragma unroll
    for (int i = 0; i < 4; ++i) {
      // A fragment: half 0 -> K {0..7,16..23}, half 1 -> K {8..15,24..31}
      const __bf16* ap = lA + (size_t)(wm + i * 16 + lrow) * LDP + hlf * 8;
      Frag16 af;
      af.u[0] = ((const uint4*)ap)[0];
      af.u[1] = ((const uint4*)(ap + 16))[0];
      #pragma unroll
      for (int j = 0; j < 2; ++j)
        acc[i][j] = __builtin_amdgcn_wmma_f32_16x16x32_bf16(
            false, af.v, false, bf[j].v, (short)0, acc[i][j], false, false);
    }
    __syncthreads();
  }

  // C layout: VGPR r, lanes 0-15 -> row m0+r, lanes 16-31 -> row m0+8+r
  #pragma unroll
  for (int i = 0; i < 4; ++i)
    #pragma unroll
    for (int j = 0; j < 2; ++j) {
      const int mbase = m0 + wm + i * 16 + hlf * 8;
      const int col = n0 + wn + j * 16 + lrow;
      #pragma unroll
      for (int r = 0; r < 8; ++r)
        gates[(size_t)(mbase + r) * H4 + col] = acc[i][j][r];
    }
}

// LSTM pointwise: gate order i,f,g,o along the 4H axis.
__global__ __launch_bounds__(256) void step_point(const float* __restrict__ gates,
                                                  const float* __restrict__ b_ih,
                                                  const float* __restrict__ b_hh,
                                                  float* __restrict__ h,
                                                  float* __restrict__ c,
                                                  __bf16* __restrict__ a_bf,
                                                  float* __restrict__ out_enc,
                                                  int t) {
  int gid = blockIdx.x * 256 + threadIdx.x;          // B*H
  int b = gid >> 8, j = gid & 255;
  const float* g = gates + (size_t)b * H4;
  float gi = fast_sigmoid(g[j]            + b_ih[j]            + b_hh[j]);
  float gf = fast_sigmoid(g[HH + j]       + b_ih[HH + j]       + b_hh[HH + j]);
  float gg = fast_tanh   (g[2 * HH + j]   + b_ih[2 * HH + j]   + b_hh[2 * HH + j]);
  float go = fast_sigmoid(g[3 * HH + j]   + b_ih[3 * HH + j]   + b_hh[3 * HH + j]);
  float cn = gf * c[gid] + gi * gg;
  float hn = go * fast_tanh(cn);
  c[gid] = cn;
  h[gid] = hn;
  a_bf[b * KC + NN + j] = (__bf16)hn;
  out_enc[(size_t)b * TM1 * HH + (size_t)t * HH + j] = hn;
}

// ---------------------------------------------------------------------------
// Host launcher
// ---------------------------------------------------------------------------
extern "C" void kernel_launch(void* const* d_in, const int* in_sizes, int n_in,
                              void* d_out, int out_size, void* d_ws, size_t ws_size,
                              hipStream_t stream) {
  const float* x    = (const float*)d_in[0];
  const float* W_ih = (const float*)d_in[1];
  const float* W_hh = (const float*)d_in[2];
  const float* b_ih = (const float*)d_in[3];
  const float* b_hh = (const float*)d_in[4];
  const float* Wc   = (const float*)d_in[5];
  const float* bc   = (const float*)d_in[6];
  const float* Wd   = (const float*)d_in[7];
  const float* bd   = (const float*)d_in[8];
  const float* Wa   = (const float*)d_in[9];
  const float* ba   = (const float*)d_in[10];

  char* ws = (char*)d_ws;
  size_t off = 0;
  auto alloc = [&](size_t bytes) {
    void* p = ws + off;
    off = (off + bytes + 255) & ~(size_t)255;
    return p;
  };
  __bf16* t2    = (__bf16*)alloc((size_t)BB * NN * 64 * 2);   // 33.6 MB
  __bf16* Wp    = (__bf16*)alloc((size_t)H4 * KC * 2);        //  1.6 MB
  __bf16* a_bf  = (__bf16*)alloc((size_t)BB * KC * 2);        //  0.8 MB
  float*  h     = (float*) alloc((size_t)BB * HH * 4);
  float*  c     = (float*) alloc((size_t)BB * HH * 4);
  float*  t1b   = (float*) alloc((size_t)BB * 64 * 4);
  float*  gts   = (float*) alloc((size_t)BB * H4 * 4);        //  2.1 MB

  float* out_w    = (float*)d_out;                            // [B,T-1,N]
  float* out_enc  = out_w   + (size_t)BB * TM1 * NN;          // [B,T-1,H]
  float* out_attn = out_enc + (size_t)BB * TM1 * HH;          // [B,T-1,N]

  pack_weights<<<(H4 * KC) / 256, 256, 0, stream>>>(W_ih, W_hh, Wp);
  init_state<<<(BB * HH) / 256, 256, 0, stream>>>(h, c, a_bf);
  compute_t2<<<BB * NN, 64, 0, stream>>>(x, Wd, bd, t2);

  for (int t = 0; t < TM1; ++t) {
    step_t1<<<(BB * 64) / 256, 256, 0, stream>>>(h, c, Wc, bc, t1b);
    step_attn<<<BB, 256, 0, stream>>>(x, t2, t1b, Wa, ba, out_w, out_attn, a_bf, t);
    step_gates_wmma<<<dim3(H4 / GBN, BB / GBM), 256, 0, stream>>>(a_bf, Wp, gts);
    step_point<<<(BB * HH) / 256, 256, 0, stream>>>(gts, b_ih, b_hh, h, c, a_bf, out_enc, t);
  }
}
